// SemanticRelation_67731634258420
// MI455X (gfx1250) — matrix operand order repaired and checked by reading, not compiled, not run on hardware
//
#include <hip/hip_runtime.h>
#include <hip/hip_bf16.h>
#include <cstdint>

// Problem dims (match reference)
#define NOBJ 256
#define DIM  1024
#define CDIM 256
#define HDIM 1024
#define ODIM 1024
#define RDIM 51
#define EDGES (NOBJ * NOBJ)
#define EPSV 1e-5f
#define SLOPE 0.01f
#define WRA_N 64   // padded [W_rel | w_att | 0...] columns

typedef __attribute__((ext_vector_type(16))) __bf16 v16bf;
typedef __attribute__((ext_vector_type(8)))  float  v8f;

union Frag16 {
    v16bf v;
    uint4  q[2];
    unsigned int u[8];
    __bf16 h[16];
};

__device__ __forceinline__ unsigned int pack_bf16(float a, float b) {
    union { unsigned int u; __bf16 h[2]; } p;
    p.h[0] = (__bf16)a;
    p.h[1] = (__bf16)b;
    return p.u;
}

// A fragment (16x32 bf16) from a row-major bf16 row pointer; two b128 loads.
// Row m's K values: lane<16 -> K {k0..k0+7, k0+16..k0+23}; lane>=16 -> +8.
__device__ __forceinline__ void load_a_bf16(Frag16& a, const unsigned int* Arow,
                                            int k0, int kbA) {
    const uint4* p = (const uint4*)(Arow + ((k0 + kbA) >> 1));
    a.q[0] = p[0];   // K = k0+kbA    .. +7
    a.q[1] = p[2];   // K = k0+kbA+16 .. +23
}

// A fragment from fp32 row (convert to bf16 in registers); four b128 loads.
__device__ __forceinline__ void load_a_f32(Frag16& a, const float* Arow,
                                           int k0, int kbA) {
    const float4* p = (const float4*)(Arow + k0 + kbA);
    float4 f0 = p[0], f1 = p[1];   // K = k0+kbA .. +7
    float4 f2 = p[4], f3 = p[5];   // K = k0+kbA+16 .. +23
    a.u[0] = pack_bf16(f0.x, f0.y);  a.u[1] = pack_bf16(f0.z, f0.w);
    a.u[2] = pack_bf16(f1.x, f1.y);  a.u[3] = pack_bf16(f1.z, f1.w);
    a.u[4] = pack_bf16(f2.x, f2.y);  a.u[5] = pack_bf16(f2.z, f2.w);
    a.u[6] = pack_bf16(f3.x, f3.y);  a.u[7] = pack_bf16(f3.z, f3.w);
}

// B fragment from pre-packed layout: Bp[((nt*kslabs + ks)*32 + lane)*16 + t].
__device__ __forceinline__ void load_b_packed(Frag16& b, const __bf16* Bp,
                                              size_t slot) {
    const uint4* p = (const uint4*)(Bp + (slot << 4));
    b.q[0] = p[0];
    b.q[1] = p[1];
}

// CDNA5 async global -> LDS copy (GLOBAL_LOAD_ASYNC_TO_LDS_B128, ASYNCcnt).
// dst is the per-lane LDS byte address (flat LDS addr truncated to 32 bits,
// per ISA 10.2).
__device__ __forceinline__ void async_copy_b128(unsigned int lds_dst,
                                                const void* src) {
    asm volatile("global_load_async_to_lds_b128 %0, %1, off"
                 :: "v"(lds_dst), "v"((unsigned long long)(uintptr_t)src)
                 : "memory");
}
__device__ __forceinline__ void wait_asynccnt0() {
    asm volatile("s_wait_asynccnt 0" ::: "memory");
}

// ---------------------------------------------------------------------------
// Pack a row-major fp32 matrix B[K,N] (optionally scaled per K-row) into
// WMMA-B fragment order, bf16. Mapping per ISA 7.12.2 (32x16 bf16 B tile):
//   lane<16 holds K={0..7,16..23} of column n, lane>=16 holds K={8..15,24..31}.
// ---------------------------------------------------------------------------
__global__ void pack_b_kernel(const float* __restrict__ B,
                              __bf16* __restrict__ Bp,
                              int K, int N, const float* __restrict__ rowscale) {
    int idx = blockIdx.x * blockDim.x + threadIdx.x;
    if (idx >= K * N) return;
    const int k = idx / N;
    const int n = idx - k * N;
    float v = B[idx];
    if (rowscale) v *= rowscale[k];
    const int nt = n >> 4, nl = n & 15;
    const int ks = k >> 5, kr = k & 31;
    const int lane = nl + (((kr >> 3) & 1) ? 16 : 0);
    const int t    = (kr & 7) + ((kr >> 4) << 3);
    const int kslabs = K >> 5;
    Bp[(((size_t)(nt * kslabs + ks)) * 32 + lane) * 16 + t] = (__bf16)v;
}

// Pack [W_rel (O x 51) | w_att (O x 1) | zeros] (K=ODIM rows, N=64 cols).
__global__ void pack_wra_kernel(const float* __restrict__ W_rel,
                                const float* __restrict__ w_att,
                                __bf16* __restrict__ Wrap) {
    int idx = blockIdx.x * blockDim.x + threadIdx.x;
    if (idx >= ODIM * WRA_N) return;
    const int k = idx / WRA_N;
    const int n = idx - k * WRA_N;
    float v = (n < RDIM) ? W_rel[k * RDIM + n]
            : (n == RDIM) ? w_att[k] : 0.0f;
    const int nt = n >> 4, nl = n & 15;
    const int ks = k >> 5, kr = k & 31;
    const int lane = nl + (((kr >> 3) & 1) ? 16 : 0);
    const int t    = (kr & 7) + ((kr >> 4) << 3);
    const int kslabs = ODIM >> 5;
    Wrap[(((size_t)(nt * kslabs + ks)) * 32 + lane) * 16 + t] = (__bf16)v;
}

// Elementwise f32 -> bf16 cast (for A operands kept row-major)
__global__ void cast_bf16_kernel(const float* __restrict__ src,
                                 __bf16* __restrict__ dst, int n) {
    int idx = blockIdx.x * blockDim.x + threadIdx.x;
    if (idx < n) dst[idx] = (__bf16)src[idx];
}

// ---------------------------------------------------------------------------
// bf16 WMMA GEMM, packed-B with double-buffered async LDS staging.
// Block = 4 waves; each wave computes 32 rows x 64 cols (2 M-tiles x 4
// N-tiles = 8 WMMAs per k-slab sharing 4 B fragments). All waves share the
// same B slab: the block stages it once per k-slab via
// GLOBAL_LOAD_ASYNC_TO_LDS_B128 (4 KB, 128 threads x 2 x b128) and overlaps
// the fill of buffer (ks+1)&1 with the WMMAs on buffer ks&1.
// __launch_bounds__(128,1) lifts the VGPR cap so the 8 f32 accumulators
// (64 VGPRs) stay in registers (no inner-loop spills).
// Grid: (N/64, M/128).
// ---------------------------------------------------------------------------
__global__ void __launch_bounds__(128, 1)
gemm_bf16_kernel(const __bf16* __restrict__ A,
                 const __bf16* __restrict__ Bp,
                 float* __restrict__ Cout,
                 const float* __restrict__ bias,
                 int K, int lda, int ldc) {
    // [phase][ntile chunk][lane][32 bytes]
    __shared__ __align__(32) unsigned char smem[2][4][32][32];

    const int lane = threadIdx.x & 31;
    const int wave = threadIdx.x >> 5;
    const int m0   = (blockIdx.y * 4 + wave) * 32;   // 2 M-tiles per wave
    const int nt0  = blockIdx.x * 4;
    const int kslabs = K >> 5;

    const int kbA = (lane < 16) ? 0 : 8;
    const unsigned int* Arow0 =
        (const unsigned int*)(A + (size_t)(m0 + (lane & 15)) * lda);
    const unsigned int* Arow1 =
        (const unsigned int*)(A + (size_t)(m0 + 16 + (lane & 15)) * lda);

    // Per-thread async stage: copy 32 bytes of chunk (nt0+wave), k-slab ks.
    const char* Bbase = (const char*)Bp;
    auto issue_stage = [&](int ks, int phase) {
        const char* src = Bbase +
            ((size_t)(nt0 + wave) * kslabs + ks) * 1024 + (size_t)lane * 32;
        unsigned int dst =
            (unsigned int)(uintptr_t)&smem[phase][wave][lane][0];
        async_copy_b128(dst,      src);
        async_copy_b128(dst + 16, src + 16);
    };

    issue_stage(0, 0);
    Frag16 a0c, a1c;
    load_a_bf16(a0c, Arow0, 0, kbA);
    load_a_bf16(a1c, Arow1, 0, kbA);

    v8f acc[8] = {};   // acc[mt*4 + nt]
    for (int ks = 0; ks < kslabs; ++ks) {
        const int phase = ks & 1;
        wait_asynccnt0();        // buffer `phase` fully written (own chunk)
        __syncthreads();         // ...and everyone else's chunk too

        if (ks + 1 < kslabs) {
            issue_stage(ks + 1, phase ^ 1);   // overlaps WMMAs below
        }

        Frag16 b0, b1, b2, b3;
        const uint4* ldsb = (const uint4*)&smem[phase][0][lane][0];
        // chunk stride in uint4s: 32*32/16 = 64
        b0.q[0] = ldsb[0];    b0.q[1] = ldsb[1];
        b1.q[0] = ldsb[64];   b1.q[1] = ldsb[65];
        b2.q[0] = ldsb[128];  b2.q[1] = ldsb[129];
        b3.q[0] = ldsb[192];  b3.q[1] = ldsb[193];

        acc[0] = __builtin_amdgcn_wmma_f32_16x16x32_bf16(false, a0c.v, false, b0.v, (short)0, acc[0], false, false);
        acc[1] = __builtin_amdgcn_wmma_f32_16x16x32_bf16(false, a0c.v, false, b1.v, (short)0, acc[1], false, false);
        acc[2] = __builtin_amdgcn_wmma_f32_16x16x32_bf16(false, a0c.v, false, b2.v, (short)0, acc[2], false, false);
        acc[3] = __builtin_amdgcn_wmma_f32_16x16x32_bf16(false, a0c.v, false, b3.v, (short)0, acc[3], false, false);
        acc[4] = __builtin_amdgcn_wmma_f32_16x16x32_bf16(false, a1c.v, false, b0.v, (short)0, acc[4], false, false);
        acc[5] = __builtin_amdgcn_wmma_f32_16x16x32_bf16(false, a1c.v, false, b1.v, (short)0, acc[5], false, false);
        acc[6] = __builtin_amdgcn_wmma_f32_16x16x32_bf16(false, a1c.v, false, b2.v, (short)0, acc[6], false, false);
        acc[7] = __builtin_amdgcn_wmma_f32_16x16x32_bf16(false, a1c.v, false, b3.v, (short)0, acc[7], false, false);

        if (ks + 1 < kslabs) {
            // a0c/a1c are dead after the WMMAs above: overwrite in place
            // (loads overlap next iteration's async-wait + barrier).
            load_a_bf16(a0c, Arow0, (ks + 1) << 5, kbA);
            load_a_bf16(a1c, Arow1, (ks + 1) << 5, kbA);
        }
    }

    // D layout: lane holds column (lane&15) of tile, rows mbase..mbase+7.
    const int nl = lane & 15;
    #pragma unroll
    for (int mt = 0; mt < 2; ++mt) {
        const int mbase = m0 + mt * 16 + ((lane >> 4) << 3);
        #pragma unroll
        for (int t = 0; t < 4; ++t) {
            const int nc = (nt0 + t) * 16 + nl;
            const float bv = bias ? bias[nc] : 0.0f;
            #pragma unroll
            for (int r = 0; r < 8; ++r)
                Cout[(size_t)(mbase + r) * ldc + nc] = acc[mt * 4 + t][r] + bv;
        }
    }
}

// ---------------------------------------------------------------------------
// Pass 1: h_act = LeakyReLU(cemb @ W1c + hi[i] + hj[j] + b1), bf16 out;
// per-channel sum/sumsq for BatchNorm. Wave: 16 edges x 64 channels.
// Grid: (HDIM/64, EDGES/64), block 128.
// ---------------------------------------------------------------------------
__global__ void pass1_kernel(const float* __restrict__ cemb,
                             const __bf16* __restrict__ W1cp,
                             const float* __restrict__ hi,
                             const float* __restrict__ hj,
                             const float* __restrict__ b1,
                             __bf16* __restrict__ hact,
                             float* __restrict__ sums,
                             float* __restrict__ sumsq) {
    const int lane = threadIdx.x & 31;
    const int wave = threadIdx.x >> 5;
    const int e0   = (blockIdx.y * 4 + wave) * 16;
    const int nt0  = blockIdx.x * 4;
    const int kslabs = CDIM >> 5;   // 8

    const int mA  = e0 + (lane & 15);
    const int kbA = (lane < 16) ? 0 : 8;
    const float* Arow = cemb + (size_t)mA * CDIM;

    v8f acc0 = {}, acc1 = {}, acc2 = {}, acc3 = {};
    #pragma unroll
    for (int ks = 0; ks < kslabs; ++ks) {
        Frag16 a, b0, b1f, b2, b3;
        load_a_f32(a, Arow, ks << 5, kbA);
        load_b_packed(b0,  W1cp, (size_t)((nt0 + 0) * kslabs + ks) * 32 + lane);
        load_b_packed(b1f, W1cp, (size_t)((nt0 + 1) * kslabs + ks) * 32 + lane);
        load_b_packed(b2,  W1cp, (size_t)((nt0 + 2) * kslabs + ks) * 32 + lane);
        load_b_packed(b3,  W1cp, (size_t)((nt0 + 3) * kslabs + ks) * 32 + lane);
        acc0 = __builtin_amdgcn_wmma_f32_16x16x32_bf16(false, a.v, false, b0.v,  (short)0, acc0, false, false);
        acc1 = __builtin_amdgcn_wmma_f32_16x16x32_bf16(false, a.v, false, b1f.v, (short)0, acc1, false, false);
        acc2 = __builtin_amdgcn_wmma_f32_16x16x32_bf16(false, a.v, false, b2.v,  (short)0, acc2, false, false);
        acc3 = __builtin_amdgcn_wmma_f32_16x16x32_bf16(false, a.v, false, b3.v,  (short)0, acc3, false, false);
    }

    const int mbase = e0 + ((lane >> 4) << 3);
    const int nl    = lane & 15;
    v8f* accs[4] = { &acc0, &acc1, &acc2, &acc3 };
    #pragma unroll
    for (int t = 0; t < 4; ++t) {
        const int h  = (nt0 + t) * 16 + nl;
        const float bb = b1[h];
        float s = 0.0f, s2 = 0.0f;
        #pragma unroll
        for (int r = 0; r < 8; ++r) {
            const int e = mbase + r;
            const int i = e >> 8;
            const int j = e & (NOBJ - 1);
            float val = (*accs[t])[r] + hi[i * HDIM + h] + hj[j * HDIM + h] + bb;
            val = (val >= 0.0f) ? val : SLOPE * val;
            hact[(size_t)e * HDIM + h] = (__bf16)val;
            s  += val;
            s2 += val * val;
        }
        atomicAdd(&sums[h], s);
        atomicAdd(&sumsq[h], s2);
    }
}

// ---------------------------------------------------------------------------
// BN stats -> per-channel scale/shift (folded into W2 by pack_b_kernel)
// ---------------------------------------------------------------------------
__global__ void stats_kernel(const float* __restrict__ sums,
                             const float* __restrict__ sumsq,
                             const float* __restrict__ gamma,
                             const float* __restrict__ beta,
                             float* __restrict__ scale,
                             float* __restrict__ shift) {
    int h = blockIdx.x * blockDim.x + threadIdx.x;
    if (h >= HDIM) return;
    const float inv_n = 1.0f / (float)EDGES;
    float mean = sums[h] * inv_n;
    float var  = sumsq[h] * inv_n - mean * mean;
    float sc   = gamma[h] * rsqrtf(var + EPSV);
    scale[h] = sc;
    shift[h] = beta[h] - mean * sc;
}

__global__ void cvec_kernel(const float* __restrict__ W2,
                            const float* __restrict__ shift,
                            const float* __restrict__ b2,
                            float* __restrict__ cvec) {
    int o = blockIdx.x * blockDim.x + threadIdx.x;
    if (o >= ODIM) return;
    float acc = b2[o];
    for (int hh = 0; hh < HDIM; ++hh)
        acc += shift[hh] * W2[(size_t)hh * ODIM + o];
    cvec[o] = acc;
}

// ---------------------------------------------------------------------------
// Pass 3: [rel | att_logit] = feat @ [W_rel|w_att] (feat fp32 -> bf16 on load)
// Wave: 16 edges x all 64 cols. Block = 4 waves = 64 edges. Grid: EDGES/64.
// ---------------------------------------------------------------------------
__global__ void pass3_kernel(const float* __restrict__ feat,
                             const __bf16* __restrict__ Wrap,
                             const float* __restrict__ b_rel,
                             const float* __restrict__ b_att,
                             float* __restrict__ rel,
                             float* __restrict__ confl) {
    const int lane = threadIdx.x & 31;
    const int wave = threadIdx.x >> 5;
    const int e0   = (blockIdx.x * 4 + wave) * 16;
    const int kslabs = ODIM >> 5;   // 32

    const int mA  = e0 + (lane & 15);
    const int kbA = (lane < 16) ? 0 : 8;
    const float* Arow = feat + (size_t)mA * ODIM;

    v8f acc0 = {}, acc1 = {}, acc2 = {}, acc3 = {};
    for (int ks = 0; ks < kslabs; ++ks) {
        Frag16 a, b0, b1, b2, b3;
        load_a_f32(a, Arow, ks << 5, kbA);
        __builtin_prefetch(Arow + ((ks + 1) << 5), 0, 1);
        load_b_packed(b0, Wrap, (size_t)(0 * kslabs + ks) * 32 + lane);
        load_b_packed(b1, Wrap, (size_t)(1 * kslabs + ks) * 32 + lane);
        load_b_packed(b2, Wrap, (size_t)(2 * kslabs + ks) * 32 + lane);
        load_b_packed(b3, Wrap, (size_t)(3 * kslabs + ks) * 32 + lane);
        acc0 = __builtin_amdgcn_wmma_f32_16x16x32_bf16(false, a.v, false, b0.v, (short)0, acc0, false, false);
        acc1 = __builtin_amdgcn_wmma_f32_16x16x32_bf16(false, a.v, false, b1.v, (short)0, acc1, false, false);
        acc2 = __builtin_amdgcn_wmma_f32_16x16x32_bf16(false, a.v, false, b2.v, (short)0, acc2, false, false);
        acc3 = __builtin_amdgcn_wmma_f32_16x16x32_bf16(false, a.v, false, b3.v, (short)0, acc3, false, false);
    }

    const int mbase = e0 + ((lane >> 4) << 3);
    const int nl    = lane & 15;
    v8f* accs[4] = { &acc0, &acc1, &acc2, &acc3 };
    #pragma unroll
    for (int t = 0; t < 4; ++t) {
        const int col = t * 16 + nl;
        if (col < RDIM) {
            const float bv = b_rel[col];
            #pragma unroll
            for (int r = 0; r < 8; ++r)
                rel[(size_t)(mbase + r) * RDIM + col] = (*accs[t])[r] + bv;
        } else if (col == RDIM) {
            const float ba = b_att[0];
            #pragma unroll
            for (int r = 0; r < 8; ++r)
                confl[mbase + r] = (*accs[t])[r] + ba;
        }
    }
}

// ---------------------------------------------------------------------------
// Row softmax over j (256 entries), in place on confl.
// ---------------------------------------------------------------------------
__global__ void softmax_kernel(float* __restrict__ confl) {
    __shared__ float red[NOBJ];
    const int i = blockIdx.x;
    const int t = threadIdx.x;
    float x = confl[i * NOBJ + t];
    red[t] = x;
    __syncthreads();
    for (int s = NOBJ / 2; s > 0; s >>= 1) {
        if (t < s) red[t] = fmaxf(red[t], red[t + s]);
        __syncthreads();
    }
    const float mx = red[0];
    __syncthreads();
    const float e = __expf(x - mx);
    red[t] = e;
    __syncthreads();
    for (int s = NOBJ / 2; s > 0; s >>= 1) {
        if (t < s) red[t] += red[t + s];
        __syncthreads();
    }
    confl[i * NOBJ + t] = e / red[0];
}

// ---------------------------------------------------------------------------
// enhanced[i, :] = sum_j conf[i, j] * feat[i, j, :]  (coalesced over O)
// ---------------------------------------------------------------------------
__global__ void enhanced_kernel(const float* __restrict__ feat,
                                const float* __restrict__ conf,
                                float* __restrict__ enh) {
    __shared__ float w[NOBJ];
    const int i = blockIdx.x;
    const int t = threadIdx.x;   // 1024 threads, one output channel each
    if (t < NOBJ) w[t] = conf[i * NOBJ + t];
    __syncthreads();
    float acc = 0.0f;
    for (int j = 0; j < NOBJ; ++j)
        acc += w[j] * feat[((size_t)(i * NOBJ + j)) * ODIM + t];
    enh[(size_t)i * ODIM + t] = acc;
}

// ---------------------------------------------------------------------------
// Host side
// ---------------------------------------------------------------------------
extern "C" void kernel_launch(void* const* d_in, const int* in_sizes, int n_in,
                              void* d_out, int out_size, void* d_ws, size_t ws_size,
                              hipStream_t stream) {
    const float* obj   = (const float*)d_in[0];   // [N, D]
    const float* cemb  = (const float*)d_in[1];   // [N, N, C]
    const float* W1    = (const float*)d_in[2];   // [2D+C, H]
    const float* b1    = (const float*)d_in[3];   // [H]
    const float* gamma = (const float*)d_in[4];   // [H]
    const float* beta  = (const float*)d_in[5];   // [H]
    const float* W2    = (const float*)d_in[6];   // [H, O]
    const float* b2    = (const float*)d_in[7];   // [O]
    const float* w_att = (const float*)d_in[8];   // [O, 1]
    const float* b_att = (const float*)d_in[9];   // [1]
    const float* W_rel = (const float*)d_in[10];  // [O, R]
    const float* b_rel = (const float*)d_in[11];  // [R]

    float* feat = (float*)d_out;                                       // [E, O]
    float* enh  = feat + (size_t)EDGES * ODIM;                         // [N, O]
    float* rel  = enh + (size_t)NOBJ * ODIM;                           // [E, R]

    // Workspace layout (all offsets 256B aligned), ~137 MB total
    char* ws = (char*)d_ws;
    size_t off = 0;
    __bf16* obj_bf  = (__bf16*)(ws + off); off += (size_t)NOBJ * DIM * 2;      // 512 KB
    __bf16* W1a_p   = (__bf16*)(ws + off); off += (size_t)DIM * HDIM * 2;      // 2 MB
    __bf16* W1b_p   = (__bf16*)(ws + off); off += (size_t)DIM * HDIM * 2;      // 2 MB
    __bf16* W1c_p   = (__bf16*)(ws + off); off += (size_t)CDIM * HDIM * 2;     // 512 KB
    __bf16* W2s_p   = (__bf16*)(ws + off); off += (size_t)HDIM * ODIM * 2;     // 2 MB
    __bf16* Wra_p   = (__bf16*)(ws + off); off += (size_t)ODIM * WRA_N * 2;    // 128 KB
    float*  hi      = (float*) (ws + off); off += (size_t)NOBJ * HDIM * 4;     // 1 MB
    float*  hj      = (float*) (ws + off); off += (size_t)NOBJ * HDIM * 4;     // 1 MB
    float*  sums    = (float*) (ws + off); off += HDIM * 4;
    float*  sumsq   = (float*) (ws + off); off += HDIM * 4;
    float*  scale   = (float*) (ws + off); off += HDIM * 4;
    float*  shift   = (float*) (ws + off); off += HDIM * 4;
    float*  cvec    = (float*) (ws + off); off += ODIM * 4;
    float*  confl   = (float*) (ws + off); off += (size_t)EDGES * 4;           // 256 KB
    __bf16* hact    = (__bf16*)(ws + off); off += (size_t)EDGES * HDIM * 2;    // 128 MB
    (void)ws_size; (void)in_sizes; (void)n_in; (void)out_size;

    hipMemsetAsync(sums,  0, HDIM * sizeof(float), stream);
    hipMemsetAsync(sumsq, 0, HDIM * sizeof(float), stream);

    // Stage 0: casts / fragment packing of weights
    {
        int n = NOBJ * DIM;
        cast_bf16_kernel<<<(n + 255) / 256, 256, 0, stream>>>(obj, obj_bf, n);
        n = DIM * HDIM;
        pack_b_kernel<<<(n + 255) / 256, 256, 0, stream>>>(W1, W1a_p, DIM, HDIM, nullptr);
        pack_b_kernel<<<(n + 255) / 256, 256, 0, stream>>>(W1 + (size_t)DIM * HDIM, W1b_p, DIM, HDIM, nullptr);
        n = CDIM * HDIM;
        pack_b_kernel<<<(n + 255) / 256, 256, 0, stream>>>(W1 + (size_t)2 * DIM * HDIM, W1c_p, CDIM, HDIM, nullptr);
        n = ODIM * WRA_N;
        pack_wra_kernel<<<(n + 255) / 256, 256, 0, stream>>>(W_rel, w_att, Wra_p);
    }

    // Stage 1: hi = obj @ W1[:D], hj = obj @ W1[D:2D]   (M=256, N=1024, K=1024)
    {
        dim3 grid(HDIM / 64, NOBJ / 128);
        gemm_bf16_kernel<<<grid, 128, 0, stream>>>(obj_bf, W1a_p, hi, nullptr, DIM, DIM, HDIM);
        gemm_bf16_kernel<<<grid, 128, 0, stream>>>(obj_bf, W1b_p, hj, nullptr, DIM, DIM, HDIM);
    }

    // Stage 2: h_act + BN statistics
    {
        dim3 grid(HDIM / 64, EDGES / 64);
        pass1_kernel<<<grid, 128, 0, stream>>>(cemb, W1c_p, hi, hj, b1, hact, sums, sumsq);
    }

    // Stage 3: fold BatchNorm into W2 (scale fused into fragment pack)
    stats_kernel<<<HDIM / 256, 256, 0, stream>>>(sums, sumsq, gamma, beta, scale, shift);
    pack_b_kernel<<<(HDIM * ODIM + 255) / 256, 256, 0, stream>>>(W2, W2s_p, HDIM, ODIM, scale);
    cvec_kernel<<<ODIM / 256, 256, 0, stream>>>(W2, shift, b2, cvec);

    // Stage 4: feat = h_act @ (scale*W2) + c   (M=65536, N=1024, K=1024)
    {
        dim3 grid(ODIM / 64, EDGES / 128);
        gemm_bf16_kernel<<<grid, 128, 0, stream>>>(hact, W2s_p, feat, cvec, HDIM, HDIM, ODIM);
    }

    // Stage 5: [relation_cls | attention logits] = feat @ [W_rel | w_att]
    pass3_kernel<<<EDGES / 64, 128, 0, stream>>>(feat, Wra_p, b_rel, b_att, rel, confl);

    // Stage 6: softmax over j, then enhanced = conf-weighted sum of feat
    softmax_kernel<<<NOBJ, NOBJ, 0, stream>>>(confl);
    enhanced_kernel<<<NOBJ, ODIM, 0, stream>>>(feat, confl, enh);
}
